// ManyToOneLSTM_18519898980509
// MI455X (gfx1250) — matrix operand order repaired and checked by reading, not compiled, not run on hardware
//
#include <hip/hip_runtime.h>
#include <stdint.h>

// ---------------------------------------------------------------------------
// Many-to-one 2-layer LSTM (H1=32, H2=64, NIN=1, NOUT=1), N=2048, S=512.
// gfx1250 wave32 strategy:
//   * Each wave owns a 16-sample batch tile for the whole 512-step sequence.
//   * Gates computed transposed:  G(4H x 16) = W(4H x H) * h^T(H x 16)
//     via v_wmma_f32_16x16x32_f16, f32 accumulation.  56 WMMAs / step / wave.
//   * Weights pre-packed once into LDS in A-operand order k = 16*hi + e (same as B)
//     (per-lane 32B contiguous -> two ds_load_b128 per tile).  A per-step
//     asm memory clobber stops LLVM from hoisting the loop-invariant LDS
//     loads into (spilled) registers -- weights must stream from LDS.
//   * h/c state lives in registers in the WMMA C/D accumulator layout;
//     accumulator->B-operand conversion is lane-local f16 packing + one
//     __shfl_xor(...,16) half-wave swap.
//   * Activations use v_exp_f32 + v_rcp_f32 (no IEEE division sequences).
// ---------------------------------------------------------------------------

typedef __attribute__((ext_vector_type(16))) _Float16 v16h;
typedef __attribute__((ext_vector_type(8)))  float    v8f;
typedef __attribute__((ext_vector_type(4)))  float    v4f;
typedef __attribute__((ext_vector_type(4)))  uint32_t v4u;

#define SEQ  512
#define H1C  32
#define H2C  64

__device__ __forceinline__ float frcp_(float x) { return __builtin_amdgcn_rcpf(x); }
__device__ __forceinline__ float fsig(float x) {
  return frcp_(1.0f + __expf(-x));                    // v_exp + v_rcp
}
__device__ __forceinline__ float ftanh_(float x) {
  float e = __expf(-2.0f * fabsf(x));
  float t = (1.0f - e) * frcp_(1.0f + e);
  return __builtin_copysignf(t, x);                   // v_bfi
}
__device__ __forceinline__ uint32_t pk2(float a, float b) {
  union { _Float16 h[2]; uint32_t u; } u;
  u.h[0] = (_Float16)a; u.h[1] = (_Float16)b;
  return u.u;
}

// Build a 32-K B operand (K x 16, f16) from two accumulator-layout f32 tiles.
// Acc tile ht: lane holds k = 16*ht + r + 8*hi for its 8 regs (col = lane%16).
// B operand:   lane holds k = 16*hi + e (e = half-element 0..15, col = lane%16)
//   -> elements 0..7  : lo lanes own t0, hi lanes need partner's t1
//   -> elements 8..15 : lo lanes need partner's t0, hi lanes own t1
__device__ __forceinline__ v16h buildB(const float* t0, const float* t1, int hi) {
  uint32_t p0[4], p1[4];
#pragma unroll
  for (int j = 0; j < 4; j++) {
    p0[j] = pk2(t0[2*j], t0[2*j+1]);
    p1[j] = pk2(t1[2*j], t1[2*j+1]);
  }
  union { uint32_t u[8]; v16h h; } b;
#pragma unroll
  for (int j = 0; j < 4; j++) {
    uint32_t c0 = (uint32_t)__shfl_xor((int)p0[j], 16, 32);
    uint32_t c1 = (uint32_t)__shfl_xor((int)p1[j], 16, 32);
    b.u[j]     = hi ? c1    : p0[j];
    b.u[4 + j] = hi ? p1[j] : c0;
  }
  return b.h;
}

// Load one pre-packed A tile (16x32 f16) from LDS: 32 contiguous bytes/lane.
__device__ __forceinline__ v16h loadA(const uint32_t* p) {
  union { v4u q[2]; v16h h; } t;
  t.q[0] = *(const v4u*)(p);
  t.q[1] = *(const v4u*)(p + 4);
  return t.h;
}
__device__ __forceinline__ v8f loadF8(const float* p) {
  union { v4f q[2]; v8f v; } t;
  t.q[0] = *(const v4f*)(p);
  t.q[1] = *(const v4f*)(p + 4);
  return t.v;
}

// Cooperatively pack W (rowTiles*16 x K, f32 row-major) into a WMMA
// A-operand layout in LDS (contiguous k, same order as buildB):
//   entry idx = a*256 + lane*8 + slot ; a = rowTile*kTiles + kTile
//   slot s : K = 2s + 16*hi (low f16), K+1 (high f16)
__device__ void fillA(uint32_t* dst, const float* W, int rowTiles, int kTiles,
                      int K, int tid, int nt) {
  int total = rowTiles * kTiles * 256;
  for (int idx = tid; idx < total; idx += nt) {
    int a = idx >> 8, rem = idx & 255;
    int l = rem >> 3, s = rem & 7;
    int rowTile = a / kTiles, kTile = a - rowTile * kTiles;
    int m = l & 15, hig = l >> 4;
    int kl = 2 * s + hig * 16;
    int row = rowTile * 16 + m;
    int kc  = kTile * 32 + kl;
    dst[idx] = pk2(W[row * K + kc], W[row * K + kc + 1]);
  }
}

#define WMMA(A, B, C) \
  __builtin_amdgcn_wmma_f32_16x16x32_f16(false, (A), false, (B), (short)0, (C), false, false)

__global__ __launch_bounds__(256) void lstm_m2o_kernel(
    const float* __restrict__ inputs,
    const float* __restrict__ h1_0, const float* __restrict__ c1_0,
    const float* __restrict__ h2_0, const float* __restrict__ c2_0,
    const float* __restrict__ Wih1, const float* __restrict__ Whh1,
    const float* __restrict__ bih1, const float* __restrict__ bhh1,
    const float* __restrict__ Wih2, const float* __restrict__ Whh2,
    const float* __restrict__ bih2, const float* __restrict__ bhh2,
    const float* __restrict__ Wlin, const float* __restrict__ blin,
    float* __restrict__ out) {
  // 56 KB packed f16 weights + 2 KB fused bias/input-column data.
  __shared__ __align__(32) uint32_t sWhh1[8  * 256];   // 8 A-tiles  (128 x 32)
  __shared__ __align__(32) uint32_t sWih2[16 * 256];   // 16 A-tiles (256 x 32)
  __shared__ __align__(32) uint32_t sWhh2[32 * 256];   // 32 A-tiles (256 x 64)
  __shared__ __align__(32) float sB1[128];             // bih1+bhh1 (gate order == index)
  __shared__ __align__(32) float sW1c[128];            // Wih1[:,0]
  __shared__ __align__(32) float sB2[256];             // bih2+bhh2

  const int tid = threadIdx.x, nt = blockDim.x;
  fillA(sWhh1, Whh1, 8, 1, 32, tid, nt);
  fillA(sWih2, Wih2, 16, 1, 32, tid, nt);
  fillA(sWhh2, Whh2, 16, 2, 64, tid, nt);
  for (int i = tid; i < 128; i += nt) { sB1[i] = bih1[i] + bhh1[i]; sW1c[i] = Wih1[i]; }
  for (int i = tid; i < 256; i += nt) { sB2[i] = bih2[i] + bhh2[i]; }
  __syncthreads();

  const int lane  = tid & 31, wave = tid >> 5;
  const int hi    = lane >> 4, col = lane & 15;
  const int tileN = blockIdx.x * (nt >> 5) + wave;     // 0..127 (16 blocks * 8 waves)
  const int n     = tileN * 16 + col;                  // batch row this lane owns
  const int rbase = hi * 8;
  const uint32_t laneOff = (uint32_t)lane * 8u;

  // Persistent state in accumulator layout: [tile][reg] -> k = 16*tile + rbase + r
  float c1s[2][8], c2s[4][8], h2v[4][8], h1t[2][8];
#pragma unroll
  for (int ht = 0; ht < 2; ht++)
#pragma unroll
    for (int r = 0; r < 8; r++) {
      int k = ht * 16 + rbase + r;
      c1s[ht][r] = c1_0[n * H1C + k];
      h1t[ht][r] = h1_0[n * H1C + k];
    }
  v16h h1B = buildB(h1t[0], h1t[1], hi);
#pragma unroll
  for (int ht = 0; ht < 4; ht++)
#pragma unroll
    for (int r = 0; r < 8; r++) {
      int k = ht * 16 + rbase + r;
      c2s[ht][r] = c2_0[n * H2C + k];
      h2v[ht][r] = h2_0[n * H2C + k];
    }
  v16h h2Ba = buildB(h2v[0], h2v[1], hi);
  v16h h2Bb = buildB(h2v[2], h2v[3], hi);

  for (int t = 0; t < SEQ; t++) {
    // Forbid hoisting/CSE of the (loop-invariant) LDS weight loads: weights
    // must stream from LDS each step instead of being spilled to scratch.
    asm volatile("" ::: "memory");

    float x = inputs[(size_t)n * SEQ + (size_t)t];

    // ---- layer 1: gates1 = Whh1 * h1^T + (bias + x*Wih1col) --------------
#pragma unroll
    for (int ht = 0; ht < 2; ht++) {
      v8f ai = loadF8(sB1 + (ht    ) * 16 + rbase) + loadF8(sW1c + (ht    ) * 16 + rbase) * x;
      v8f af = loadF8(sB1 + (2 + ht) * 16 + rbase) + loadF8(sW1c + (2 + ht) * 16 + rbase) * x;
      v8f ag = loadF8(sB1 + (4 + ht) * 16 + rbase) + loadF8(sW1c + (4 + ht) * 16 + rbase) * x;
      v8f ao = loadF8(sB1 + (6 + ht) * 16 + rbase) + loadF8(sW1c + (6 + ht) * 16 + rbase) * x;
      ai = WMMA(loadA(sWhh1 + (ht    ) * 256 + laneOff), h1B, ai);
      af = WMMA(loadA(sWhh1 + (2 + ht) * 256 + laneOff), h1B, af);
      ag = WMMA(loadA(sWhh1 + (4 + ht) * 256 + laneOff), h1B, ag);
      ao = WMMA(loadA(sWhh1 + (6 + ht) * 256 + laneOff), h1B, ao);
#pragma unroll
      for (int r = 0; r < 8; r++) {
        float c = fsig(af[r]) * c1s[ht][r] + fsig(ai[r]) * ftanh_(ag[r]);
        c1s[ht][r] = c;
        h1t[ht][r] = fsig(ao[r]) * ftanh_(c);
      }
    }
    h1B = buildB(h1t[0], h1t[1], hi);

    // ---- layer 2: gates2 = Wih2 * h1^T + Whh2 * h2^T + bias --------------
#pragma unroll
    for (int ht = 0; ht < 4; ht++) {
      v8f ai = loadF8(sB2 + (ht     ) * 16 + rbase);
      v8f af = loadF8(sB2 + (4 + ht ) * 16 + rbase);
      v8f ag = loadF8(sB2 + (8 + ht ) * 16 + rbase);
      v8f ao = loadF8(sB2 + (12 + ht) * 16 + rbase);
      ai = WMMA(loadA(sWih2 + (ht     ) * 256 + laneOff), h1B, ai);
      af = WMMA(loadA(sWih2 + (4 + ht ) * 256 + laneOff), h1B, af);
      ag = WMMA(loadA(sWih2 + (8 + ht ) * 256 + laneOff), h1B, ag);
      ao = WMMA(loadA(sWih2 + (12 + ht) * 256 + laneOff), h1B, ao);
      ai = WMMA(loadA(sWhh2 + ((ht     ) * 2    ) * 256 + laneOff), h2Ba, ai);
      ai = WMMA(loadA(sWhh2 + ((ht     ) * 2 + 1) * 256 + laneOff), h2Bb, ai);
      af = WMMA(loadA(sWhh2 + ((4 + ht ) * 2    ) * 256 + laneOff), h2Ba, af);
      af = WMMA(loadA(sWhh2 + ((4 + ht ) * 2 + 1) * 256 + laneOff), h2Bb, af);
      ag = WMMA(loadA(sWhh2 + ((8 + ht ) * 2    ) * 256 + laneOff), h2Ba, ag);
      ag = WMMA(loadA(sWhh2 + ((8 + ht ) * 2 + 1) * 256 + laneOff), h2Bb, ag);
      ao = WMMA(loadA(sWhh2 + ((12 + ht) * 2    ) * 256 + laneOff), h2Ba, ao);
      ao = WMMA(loadA(sWhh2 + ((12 + ht) * 2 + 1) * 256 + laneOff), h2Bb, ao);
#pragma unroll
      for (int r = 0; r < 8; r++) {
        float c = fsig(af[r]) * c2s[ht][r] + fsig(ai[r]) * ftanh_(ag[r]);
        c2s[ht][r] = c;
        h2v[ht][r] = fsig(ao[r]) * ftanh_(c);   // safe: WMMAs above used h2Ba/h2Bb regs
      }
    }
    h2Ba = buildB(h2v[0], h2v[1], hi);
    h2Bb = buildB(h2v[2], h2v[3], hi);
  }

  // ---- final linear: out[n] = sum_k h2[n][k] * Wlin[k] + blin ------------
  float sum = 0.0f;
#pragma unroll
  for (int ht = 0; ht < 4; ht++)
#pragma unroll
    for (int r = 0; r < 8; r++)
      sum += h2v[ht][r] * Wlin[ht * 16 + rbase + r];
  float oth = __shfl_xor(sum, 16, 32);   // combine lo/hi half-wave partials
  if (hi == 0) out[n] = sum + oth + blin[0];
}

extern "C" void kernel_launch(void* const* d_in, const int* in_sizes, int n_in,
                              void* d_out, int out_size, void* d_ws, size_t ws_size,
                              hipStream_t stream) {
  (void)in_sizes; (void)n_in; (void)out_size; (void)d_ws; (void)ws_size;
  const float* inputs = (const float*)d_in[0];
  const float* h1_0   = (const float*)d_in[1];
  const float* c1_0   = (const float*)d_in[2];
  const float* h2_0   = (const float*)d_in[3];
  const float* c2_0   = (const float*)d_in[4];
  const float* Wih1   = (const float*)d_in[5];
  const float* Whh1   = (const float*)d_in[6];
  const float* bih1   = (const float*)d_in[7];
  const float* bhh1   = (const float*)d_in[8];
  const float* Wih2   = (const float*)d_in[9];
  const float* Whh2   = (const float*)d_in[10];
  const float* bih2   = (const float*)d_in[11];
  const float* bhh2   = (const float*)d_in[12];
  const float* Wlin   = (const float*)d_in[13];
  const float* blin   = (const float*)d_in[14];
  float* out = (float*)d_out;

  // 16 blocks * 8 waves = 128 waves, one 16-sample batch tile each (N=2048).
  lstm_m2o_kernel<<<dim3(16), dim3(256), 0, stream>>>(
      inputs, h1_0, c1_0, h2_0, c2_0,
      Wih1, Whh1, bih1, bhh1, Wih2, Whh2, bih2, bhh2, Wlin, blin, out);
}
